// RVQ_86457691669013
// MI455X (gfx1250) — matrix operand order repaired and checked by reading, not compiled, not run on hardware
//
#include <hip/hip_runtime.h>
#include <hip/hip_bf16.h>

// Problem constants (from reference)
#define BQ   16
#define SQ   2048
#define DQ   256           // vector dim
#define KQ   8             // stages
#define CDQ  1024          // codebook entries per stage
#define NQ   (BQ*SQ)       // 32768 rows
#define MTILE 128          // rows per workgroup
#define RSTRIDE 260        // padded floats per residual row in LDS (breaks bank pattern)
#define CBDW 132           // padded dwords (=264 bf16) per codebook column in LDS
                           // (multiple of 4 -> every B fragment stays int4-aligned)

#define SMEM_BYTES (MTILE*RSTRIDE*4 /*residual f32*/ \
                  + 128*CBDW*4      /*cb block bf16*/ \
                  + 128*4           /*col norms*/ \
                  + 128*4           /*enc per row*/ \
                  + 256*4)          /*loss reduce*/

typedef __attribute__((ext_vector_type(16))) __bf16        v16bf;
typedef __attribute__((ext_vector_type(8)))  float         v8f;
typedef __attribute__((ext_vector_type(8)))  unsigned int  v8u;

union BfPack { v16bf h; v8u u; uint4 q[2]; };

// round-to-nearest-even f32 -> bf16 (low 16 bits of result)
__device__ __forceinline__ unsigned int f2bf(float f) {
  unsigned int u = __float_as_uint(f);
  return (u + 0x7FFFu + ((u >> 16) & 1u)) >> 16;
}

// ---------------- prep kernels ----------------

__global__ void rvq_zero(float* out) {
  if (threadIdx.x < 2) out[threadIdx.x] = 0.0f;
}

// convert all codebooks f32 -> packed bf16 pairs (one uint per 2 elems)
__global__ void rvq_prep_cb(const float* __restrict__ cb,
                            unsigned int* __restrict__ cbh) {
  size_t idx = (size_t)blockIdx.x * 256 + threadIdx.x; // pair index
  float2 f = ((const float2*)cb)[idx];
  cbh[idx] = f2bf(f.x) | (f2bf(f.y) << 16);
}

// per-codebook-entry squared norm (f32, exact)
__global__ void rvq_prep_norm(const float* __restrict__ cb,
                              float* __restrict__ norms) {
  int r = blockIdx.x * 256 + threadIdx.x;   // K*CD = 8192 rows
  const float* p = cb + (size_t)r * DQ;
  float s = 0.0f;
  for (int d = 0; d < DQ; ++d) { float v = p[d]; s += v * v; }
  norms[r] = s;
}

// ---------------- main RVQ kernel ----------------

__global__ void __launch_bounds__(256, 1)
rvq_main(const float* __restrict__ x, const float* __restrict__ cb,
         const unsigned int* __restrict__ cbh, const float* __restrict__ cbnorm,
         float* __restrict__ out) {
  extern __shared__ char smem[];
  float*        resf   = (float*)smem;                             // MTILE x RSTRIDE
  unsigned int* cbu    = (unsigned int*)(smem + MTILE*RSTRIDE*4);  // 128 x CBDW dwords
  float*        nrm    = (float*)((char*)cbu + 128*CBDW*4);        // 128
  int*          enc_sh = (int*)(nrm + 128);                        // 128
  float*        red    = (float*)(enc_sh + 128);                   // 256

  const int tid   = threadIdx.x;
  const int lane  = tid & 31;
  const int wave  = tid >> 5;
  const int wg    = blockIdx.x;
  const int rowbase = wg * MTILE;
  const int col16 = lane & 15;
  const int hi8   = (lane >= 16) ? 8 : 0;

  // ---- load x tile into residual LDS (coalesced) ----
  for (int idx = tid; idx < MTILE * DQ; idx += 256) {
    int r = idx >> 8, c0 = idx & 255;
    resf[r * RSTRIDE + c0] = x[(size_t)(rowbase + r) * DQ + c0];
  }
  __syncthreads();

  const float* rrow = resf + (wave * 16 + col16) * RSTRIDE;
  float lsum = 0.0f;
  float* out_enc = out + 2;
  float* out_q   = out + 2 + (size_t)NQ * KQ;

  for (int i = 0; i < KQ; ++i) {
    // ---- rebuild bf16 A fragments from current residual (contiguous k = 16*half + s order, same as B) ----
    BfPack amat[8];
#pragma unroll
    for (int kc = 0; kc < 8; ++kc) {
#pragma unroll
      for (int j = 0; j < 8; ++j) {
        int k0 = kc * 32 + (hi8 << 1) + (j << 1);
        float2 f = *(const float2*)(rrow + k0);
        amat[kc].u[j] = f2bf(f.x) | (f2bf(f.y) << 16);
      }
    }

    float minv[8];
    int   mini[8];
#pragma unroll
    for (int v = 0; v < 8; ++v) { minv[v] = 3.0e38f; mini[v] = 0; }

    for (int blk = 0; blk < CDQ / 128; ++blk) {
      __syncthreads();   // previous compute done before overwriting cb block
      // ---- stage 128 bf16 codebook columns into LDS ----
      {
        const unsigned int* src = cbh + ((size_t)(i * CDQ + blk * 128) * (DQ / 2));
        if (tid == 0) __builtin_prefetch(src + 128 * (DQ / 2), 0, 1); // next block -> GL2
        for (int idx = tid; idx < 128 * 32; idx += 256) {   // 32 uint4 per column
          int c0 = idx >> 5, q = idx & 31;
          uint4 v4 = ((const uint4*)src)[c0 * 32 + q];
          *((uint4*)(cbu + c0 * CBDW + q * 4)) = v4;
        }
        if (tid < 128) nrm[tid] = cbnorm[i * CDQ + blk * 128 + tid];
      }
      __syncthreads();

      // ---- column tiles in pairs: two independent WMMA accumulation chains ----
#pragma unroll 1
      for (int t = 0; t < 8; t += 2) {
        v8f c0 = {0.f, 0.f, 0.f, 0.f, 0.f, 0.f, 0.f, 0.f};
        v8f c1 = {0.f, 0.f, 0.f, 0.f, 0.f, 0.f, 0.f, 0.f};
#pragma unroll
        for (int kc = 0; kc < 8; ++kc) {
          BfPack b0, b1;
          const uint4* p0 = (const uint4*)(cbu + (t * 16 + col16) * CBDW
                                           + kc * 16 + hi8);
          const uint4* p1 = (const uint4*)(cbu + ((t + 1) * 16 + col16) * CBDW
                                           + kc * 16 + hi8);
          b0.q[0] = p0[0]; b0.q[1] = p0[1];   // -> ds_load_b128 x2
          b1.q[0] = p1[0]; b1.q[1] = p1[1];
          c0 = __builtin_amdgcn_wmma_f32_16x16x32_bf16(
              false, amat[kc].h, false, b0.h, (short)0, c0, false, false);
          c1 = __builtin_amdgcn_wmma_f32_16x16x32_bf16(
              false, amat[kc].h, false, b1.h, (short)0, c1, false, false);
        }
        float cn0 = nrm[t * 16 + col16];
        float cn1 = nrm[(t + 1) * 16 + col16];
        int   cg0 = blk * 128 + t * 16 + col16;
        int   cg1 = cg0 + 16;
#pragma unroll
        for (int v = 0; v < 8; ++v) {
          float s0 = cn0 - 2.0f * c0[v];     // ||c||^2 - 2 res.c  (argmin-equivalent)
          float s1 = cn1 - 2.0f * c1[v];
          if (s0 < minv[v]) { minv[v] = s0; mini[v] = cg0; }
          if (s1 < minv[v]) { minv[v] = s1; mini[v] = cg1; }
        }
      }
    } // blk

    // ---- argmin across the 16 lanes of each C-matrix half (rows v / v+8) ----
#pragma unroll
    for (int v = 0; v < 8; ++v) {
      for (int m = 1; m < 16; m <<= 1) {
        float ov = __shfl_xor(minv[v], m, 32);
        int   oi = __shfl_xor(mini[v], m, 32);
        if (ov < minv[v] || (ov == minv[v] && oi < mini[v])) {
          minv[v] = ov; mini[v] = oi;
        }
      }
    }
    if ((lane & 15) == 0) {
      int hf = lane >> 4;
#pragma unroll
      for (int v = 0; v < 8; ++v) {
        int r = v + 8 * hf;
        enc_sh[wave * 16 + r] = mini[v];
        out_enc[(size_t)(rowbase + wave * 16 + r) * KQ + i] = (float)mini[v];
      }
    }

    // ---- residual update in exact f32 (wave-local rows) + loss accumulation ----
    for (int r = 0; r < 16; ++r) {
      int ci = enc_sh[wave * 16 + r];                   // uniform within wave
      const float* crow = cb + ((size_t)i * CDQ + ci) * DQ;
      float* rr = resf + (wave * 16 + r) * RSTRIDE;
#pragma unroll
      for (int j2 = 0; j2 < 8; ++j2) {
        int d0 = lane + 32 * j2;
        float nv = rr[d0] - crow[d0];
        rr[d0] = nv;
        lsum += nv * nv;                                 // (res - near)^2
      }
    }
  } // stages

  // ---- quantised = x - final residual ----
  for (int r = 0; r < 16; ++r) {
    int gr = rowbase + wave * 16 + r;
    const float* xr = x + (size_t)gr * DQ;
    const float* rr = resf + (wave * 16 + r) * RSTRIDE;
    float* qr = out_q + (size_t)gr * DQ;
#pragma unroll
    for (int j2 = 0; j2 < 8; ++j2) {
      int d0 = lane + 32 * j2;
      qr[d0] = xr[d0] - rr[d0];
    }
  }

  // ---- loss reduction: loss_cd == loss_enc forward ----
  red[tid] = lsum;
  __syncthreads();
  if (tid == 0) {
    float s = 0.0f;
    for (int t2 = 0; t2 < 256; ++t2) s += red[t2];
    s *= (1.0f / ((float)NQ * (float)DQ));
    atomicAdd(out + 0, s);
    atomicAdd(out + 1, s);
  }
}

// ---------------- launcher ----------------

extern "C" void kernel_launch(void* const* d_in, const int* in_sizes, int n_in,
                              void* d_out, int out_size, void* d_ws, size_t ws_size,
                              hipStream_t stream) {
  (void)in_sizes; (void)n_in; (void)out_size; (void)ws_size;
  const float* x  = (const float*)d_in[0];
  const float* cb = (const float*)d_in[1];
  float* out = (float*)d_out;

  // workspace: bf16 codebooks (4 MB) + norms (32 KB)
  unsigned int* cbh   = (unsigned int*)d_ws;
  float*        norms = (float*)((char*)d_ws + (size_t)KQ * CDQ * DQ * 2);

  rvq_zero<<<1, 64, 0, stream>>>(out);
  rvq_prep_cb<<<(KQ * CDQ * DQ / 2) / 256, 256, 0, stream>>>(cb, cbh);   // 4096 blocks
  rvq_prep_norm<<<(KQ * CDQ) / 256, 256, 0, stream>>>(cb, norms);        // 32 blocks
  rvq_main<<<NQ / MTILE, 256, SMEM_BYTES, stream>>>(x, cb, cbh, norms, out);
}
